// CLAM_SB_61125974557061
// MI455X (gfx1250) — matrix operand order repaired and checked by reading, not compiled, not run on hardware
//
#include <hip/hip_runtime.h>
#include <math.h>

#define N_BAGS 8192
#define L_IN   1024
#define H_DIM  512
#define D_DIM  256

typedef __bf16 bf16_t;
typedef __attribute__((ext_vector_type(16))) __bf16 v16bf;
typedef __attribute__((ext_vector_type(8)))  __bf16 v8bf;
typedef __attribute__((ext_vector_type(8)))  float  v8f;
typedef __attribute__((ext_vector_type(4)))  float  f32x4;

// ---------------------------------------------------------------------------
// Zero-fill d_out (logits + A_raw) with nontemporal 128-bit stores: the 256MB
// A_raw write is the HBM roofline floor; NT keeps it out of the 192MB L2 so
// the GEMM working set stays resident.
// ---------------------------------------------------------------------------
__global__ __launch_bounds__(256) void k_fill_zero(float* __restrict__ p, long long n)
{
    long long n4 = n >> 2;
    long long i = (long long)blockIdx.x * blockDim.x + threadIdx.x;
    long long stride = (long long)gridDim.x * blockDim.x;
    f32x4 z = {0.f, 0.f, 0.f, 0.f};
    for (long long j = i; j < n4; j += stride)
        __builtin_nontemporal_store(z, ((f32x4*)p) + j);
    long long base = n4 << 2;
    if (i < (n - base)) p[base + i] = 0.f;
}

// Split f32 -> (bf16 hi, bf16 lo) so bf16x3 WMMA reproduces ~fp32 accuracy.
__global__ __launch_bounds__(256) void k_split(const float* __restrict__ in,
                                               bf16_t* __restrict__ hi,
                                               bf16_t* __restrict__ lo, int n)
{
    int i = blockIdx.x * blockDim.x + threadIdx.x;
    if (i < n) {
        float v = in[i];
        bf16_t h = (bf16_t)v;
        hi[i] = h;
        lo[i] = (bf16_t)(v - (float)h);
    }
}

// Transpose [K][Nn] f32 weights -> [Nn][K] bf16 hi/lo, so each lane's WMMA
// B-fragment (a K-column) is two contiguous 16B loads.
__global__ __launch_bounds__(256) void k_tsplit(const float* __restrict__ in,
                                                bf16_t* __restrict__ hi,
                                                bf16_t* __restrict__ lo,
                                                int K, int Nn)
{
    int i = blockIdx.x * blockDim.x + threadIdx.x;
    if (i < K * Nn) {
        int k = i / Nn, n = i % Nn;
        float v = in[i];
        bf16_t h = (bf16_t)v;
        hi[n * K + k] = h;
        lo[n * K + k] = (bf16_t)(v - (float)h);
    }
}

// --- WMMA fragment loaders (per CDNA5 ISA 16-bit layouts, wave32) ----------
// A (16x32, MxK): lanes 0-15 row M=lane, elems 0..7 = K..K+7, 8..15 = K+16..23
//                 lanes 16-31 same rows, K offset +8.
__device__ __forceinline__ v16bf load_a_frag(const bf16_t* __restrict__ tile,
                                             int lda, int lane)
{
    int row = lane & 15;
    int kk  = (lane >> 4) << 3;          // 0 or 8
    const bf16_t* p = tile + row * lda + kk;
    union { v16bf v; v8bf h[2]; } u;
    u.h[0] = *(const v8bf*)(p);
    u.h[1] = *(const v8bf*)(p + 16);
    return u.v;
}
// B (32x16, KxN) from transposed weights [N][K]: lanes 0-15 col N=lane holds
// K..K+7, K+16..K+23; lanes 16-31 same with K offset +8.
__device__ __forceinline__ v16bf load_b_frag(const bf16_t* __restrict__ tileT,
                                             int ldb, int lane)
{
    int col = lane & 15;
    int kk  = (lane >> 4) << 3;          // 0 or 8
    const bf16_t* p = tileT + col * ldb + kk;
    union { v16bf v; v8bf h[2]; } u;
    u.h[0] = *(const v8bf*)(p);
    u.h[1] = *(const v8bf*)(p + 16);
    return u.v;
}

__device__ __forceinline__ v8f wmma_bf16x3(v16bf ah, v16bf al, v16bf bh, v16bf bl, v8f c)
{
    c = __builtin_amdgcn_wmma_f32_16x16x32_bf16(false, ah, false, bh, (short)0, c, false, false);
    c = __builtin_amdgcn_wmma_f32_16x16x32_bf16(false, ah, false, bl, (short)0, c, false, false);
    c = __builtin_amdgcn_wmma_f32_16x16x32_bf16(false, al, false, bh, (short)0, c, false, false);
    return c;
}

// ---------------------------------------------------------------------------
// GEMM1: h = relu(x @ W_fc + b_fc).  M=8192, K=1024, N=512.
// Block = 8 waves (4 in M x 2 in N); wave tile 32x64 -> block tile 128x128.
// ---------------------------------------------------------------------------
__global__ __launch_bounds__(256) void k_gemm1(const bf16_t* __restrict__ xhi,
                                               const bf16_t* __restrict__ xlo,
                                               const bf16_t* __restrict__ wthi,
                                               const bf16_t* __restrict__ wtlo,
                                               const float* __restrict__ bfc,
                                               float* __restrict__ h)
{
    int lane = threadIdx.x & 31;
    int w    = threadIdx.x >> 5;
    int wm = w & 3, wn = w >> 2;
    int m0 = blockIdx.x * 128 + wm * 32;
    int n0 = blockIdx.y * 128 + wn * 64;

    v8f acc[2][4];
#pragma unroll
    for (int i = 0; i < 2; i++)
#pragma unroll
        for (int j = 0; j < 4; j++) acc[i][j] = (v8f){0,0,0,0,0,0,0,0};

    for (int k = 0; k < L_IN; k += 32) {
        v16bf ah[2], al[2];
#pragma unroll
        for (int mt = 0; mt < 2; mt++) {
            ah[mt] = load_a_frag(xhi + (m0 + mt*16) * L_IN + k, L_IN, lane);
            al[mt] = load_a_frag(xlo + (m0 + mt*16) * L_IN + k, L_IN, lane);
        }
#pragma unroll
        for (int nt = 0; nt < 4; nt++) {
            v16bf bh = load_b_frag(wthi + (n0 + nt*16) * L_IN + k, L_IN, lane);
            v16bf bl = load_b_frag(wtlo + (n0 + nt*16) * L_IN + k, L_IN, lane);
#pragma unroll
            for (int mt = 0; mt < 2; mt++)
                acc[mt][nt] = wmma_bf16x3(ah[mt], al[mt], bh, bl, acc[mt][nt]);
        }
    }

    // C/D layout: lanes 0-15 col=lane rows r; lanes 16-31 col=lane-16 rows r+8
    int col_t = lane & 15;
    int rbase = (lane >> 4) << 3;
#pragma unroll
    for (int mt = 0; mt < 2; mt++)
#pragma unroll
        for (int nt = 0; nt < 4; nt++) {
            int col = n0 + nt * 16 + col_t;
            float bias = bfc[col];
#pragma unroll
            for (int r = 0; r < 8; r++) {
                int row = m0 + mt * 16 + rbase + r;
                float v = acc[mt][nt][r] + bias;
                h[row * H_DIM + col] = v > 0.f ? v : 0.f;
            }
        }
}

// ---------------------------------------------------------------------------
// GEMM2 (fused gates): ab = tanh(h@W_a+b_a) * sigmoid(h@W_b+b_b).
// Same tiling, A-fragments shared across the two weight streams.
// ---------------------------------------------------------------------------
__global__ __launch_bounds__(256) void k_gemm2(const bf16_t* __restrict__ hhi,
                                               const bf16_t* __restrict__ hlo,
                                               const bf16_t* __restrict__ wahi,
                                               const bf16_t* __restrict__ walo,
                                               const bf16_t* __restrict__ wbhi,
                                               const bf16_t* __restrict__ wblo,
                                               const float* __restrict__ ba,
                                               const float* __restrict__ bb,
                                               float* __restrict__ ab)
{
    int lane = threadIdx.x & 31;
    int w    = threadIdx.x >> 5;
    int wm = w & 3, wn = w >> 2;
    int m0 = blockIdx.x * 128 + wm * 32;
    int n0 = blockIdx.y * 128 + wn * 64;

    v8f accA[2][4], accB[2][4];
#pragma unroll
    for (int i = 0; i < 2; i++)
#pragma unroll
        for (int j = 0; j < 4; j++) {
            accA[i][j] = (v8f){0,0,0,0,0,0,0,0};
            accB[i][j] = (v8f){0,0,0,0,0,0,0,0};
        }

    for (int k = 0; k < H_DIM; k += 32) {
        v16bf ah[2], al[2];
#pragma unroll
        for (int mt = 0; mt < 2; mt++) {
            ah[mt] = load_a_frag(hhi + (m0 + mt*16) * H_DIM + k, H_DIM, lane);
            al[mt] = load_a_frag(hlo + (m0 + mt*16) * H_DIM + k, H_DIM, lane);
        }
#pragma unroll
        for (int nt = 0; nt < 4; nt++) {
            v16bf bah = load_b_frag(wahi + (n0 + nt*16) * H_DIM + k, H_DIM, lane);
            v16bf bal = load_b_frag(walo + (n0 + nt*16) * H_DIM + k, H_DIM, lane);
            v16bf bbh = load_b_frag(wbhi + (n0 + nt*16) * H_DIM + k, H_DIM, lane);
            v16bf bbl = load_b_frag(wblo + (n0 + nt*16) * H_DIM + k, H_DIM, lane);
#pragma unroll
            for (int mt = 0; mt < 2; mt++) {
                accA[mt][nt] = wmma_bf16x3(ah[mt], al[mt], bah, bal, accA[mt][nt]);
                accB[mt][nt] = wmma_bf16x3(ah[mt], al[mt], bbh, bbl, accB[mt][nt]);
            }
        }
    }

    int col_t = lane & 15;
    int rbase = (lane >> 4) << 3;
#pragma unroll
    for (int mt = 0; mt < 2; mt++)
#pragma unroll
        for (int nt = 0; nt < 4; nt++) {
            int col = n0 + nt * 16 + col_t;
            float biasA = ba[col], biasB = bb[col];
#pragma unroll
            for (int r = 0; r < 8; r++) {
                int row = m0 + mt * 16 + rbase + r;
                float va = tanhf(accA[mt][nt][r] + biasA);
                float vb = accB[mt][nt][r] + biasB;
                vb = 1.f / (1.f + expf(-vb));
                ab[row * D_DIM + col] = va * vb;
            }
        }
}

// scores[i] = ab[i,:] . W_c + b_c  -- one wave per row.
__global__ __launch_bounds__(256) void k_scores(const float* __restrict__ ab,
                                                const float* __restrict__ wc,
                                                const float* __restrict__ bc,
                                                float* __restrict__ scores)
{
    int lane = threadIdx.x & 31;
    int w    = threadIdx.x >> 5;
    int row  = blockIdx.x * 8 + w;
    const float* p = ab + row * D_DIM;
    float s = 0.f;
#pragma unroll
    for (int t = 0; t < D_DIM / 32; t++) {
        int j = lane + t * 32;
        s += p[j] * wc[j];
    }
#pragma unroll
    for (int off = 16; off; off >>= 1) s += __shfl_xor(s, off, 32);
    if (lane == 0) scores[row] = s + bc[0];
}

// Single block: m = max(max(scores),0); sumE = sum exp(scores-m).
__global__ __launch_bounds__(1024) void k_reduce(const float* __restrict__ scores,
                                                 float* __restrict__ red)
{
    __shared__ float sm[1024];
    int t = threadIdx.x;
    float mx = -3.4e38f;
    for (int i = t; i < N_BAGS; i += 1024) mx = fmaxf(mx, scores[i]);
    sm[t] = mx; __syncthreads();
    for (int s = 512; s; s >>= 1) { if (t < s) sm[t] = fmaxf(sm[t], sm[t + s]); __syncthreads(); }
    float m = fmaxf(sm[0], 0.f);
    __syncthreads();
    float se = 0.f;
    for (int i = t; i < N_BAGS; i += 1024) se += expf(scores[i] - m);
    sm[t] = se; __syncthreads();
    for (int s = 512; s; s >>= 1) { if (t < s) sm[t] += sm[t + s]; __syncthreads(); }
    if (t == 0) { red[0] = m; red[1] = sm[0]; }
}

// Mv[j] = sum_i w_i * h[i][j], w_i = (e_i + (n-1)*e0)/Z. 4 columns per block.
__global__ __launch_bounds__(256) void k_wsum(const float* __restrict__ h,
                                              const float* __restrict__ scores,
                                              const float* __restrict__ red,
                                              float* __restrict__ Mv)
{
    __shared__ float sm[4][256];
    int t  = threadIdx.x;
    int j0 = blockIdx.x * 4;
    float m = red[0], sumE = red[1];
    float e0 = expf(-m);
    float Z  = sumE + ((float)N_BAGS * (float)N_BAGS - (float)N_BAGS) * e0;
    float addw = (float)(N_BAGS - 1) * e0;
    float a0 = 0.f, a1 = 0.f, a2 = 0.f, a3 = 0.f;
    for (int i = t; i < N_BAGS; i += 256) {
        float wgt = (expf(scores[i] - m) + addw) / Z;
        const float* hp = h + i * H_DIM + j0;
        a0 += wgt * hp[0]; a1 += wgt * hp[1]; a2 += wgt * hp[2]; a3 += wgt * hp[3];
    }
    sm[0][t] = a0; sm[1][t] = a1; sm[2][t] = a2; sm[3][t] = a3;
    __syncthreads();
    for (int s = 128; s; s >>= 1) {
        if (t < s) {
#pragma unroll
            for (int c = 0; c < 4; c++) sm[c][t] += sm[c][t + s];
        }
        __syncthreads();
    }
    if (t < 4) Mv[j0 + t] = sm[t][0];
}

// logits = Mv @ W_cls + b_cls  (512x2), written to d_out[0..1].
__global__ __launch_bounds__(256) void k_logits(const float* __restrict__ Mv,
                                                const float* __restrict__ wcls,
                                                const float* __restrict__ bcls,
                                                float* __restrict__ out)
{
    __shared__ float sm[2][256];
    int t = threadIdx.x;
    float c0 = 0.f, c1 = 0.f;
    for (int j = t; j < H_DIM; j += 256) {
        float mv = Mv[j];
        c0 += mv * wcls[j * 2 + 0];
        c1 += mv * wcls[j * 2 + 1];
    }
    sm[0][t] = c0; sm[1][t] = c1; __syncthreads();
    for (int s = 128; s; s >>= 1) {
        if (t < s) { sm[0][t] += sm[0][t + s]; sm[1][t] += sm[1][t + s]; }
        __syncthreads();
    }
    if (t == 0) { out[0] = sm[0][0] + bcls[0]; out[1] = sm[1][0] + bcls[1]; }
}

__global__ __launch_bounds__(256) void k_diag(const float* __restrict__ scores,
                                              float* __restrict__ araw)
{
    int i = blockIdx.x * blockDim.x + threadIdx.x;
    if (i < N_BAGS) araw[(long long)i * (N_BAGS + 1)] = scores[i];
}

// ---------------------------------------------------------------------------
extern "C" void kernel_launch(void* const* d_in, const int* in_sizes, int n_in,
                              void* d_out, int out_size, void* d_ws, size_t ws_size,
                              hipStream_t stream)
{
    const float* x     = (const float*)d_in[0];
    const float* W_fc  = (const float*)d_in[1];
    const float* b_fc  = (const float*)d_in[2];
    const float* W_a   = (const float*)d_in[3];
    const float* b_a   = (const float*)d_in[4];
    const float* W_b   = (const float*)d_in[5];
    const float* b_b   = (const float*)d_in[6];
    const float* W_c   = (const float*)d_in[7];
    const float* b_c   = (const float*)d_in[8];
    const float* W_cls = (const float*)d_in[9];
    const float* b_cls = (const float*)d_in[10];
    float* out = (float*)d_out;

    char* ws = (char*)d_ws;
    const size_t MB = 1ull << 20;
    bf16_t* xhi     = (bf16_t*)(ws + 0 * MB);        // 16 MB
    bf16_t* xlo     = (bf16_t*)(ws + 16 * MB);       // 16 MB
    float*  h       = (float*) (ws + 32 * MB);       // 16 MB
    bf16_t* wfct_hi = (bf16_t*)(ws + 48 * MB);       // 1 MB
    bf16_t* wfct_lo = (bf16_t*)(ws + 49 * MB);       // 1 MB
    bf16_t* wat_hi  = (bf16_t*)(ws + 50 * MB);       // 256 KB
    bf16_t* wat_lo  = (bf16_t*)(ws + 50 * MB + 256 * 1024);
    bf16_t* wbt_hi  = (bf16_t*)(ws + 50 * MB + 512 * 1024);
    bf16_t* wbt_lo  = (bf16_t*)(ws + 50 * MB + 768 * 1024);
    float*  scores  = (float*) (ws + 51 * MB);       // 32 KB
    float*  red     = (float*) (ws + 51 * MB + 32 * 1024);
    float*  Mv      = (float*) (ws + 51 * MB + 40 * 1024);
    // Aliases reusing the x-split region after GEMM1 is done:
    bf16_t* hhi     = (bf16_t*)(ws + 0 * MB);        // 8 MB
    bf16_t* hlo     = (bf16_t*)(ws + 8 * MB);        // 8 MB
    float*  abuf    = (float*) (ws + 16 * MB);       // 8 MB

    const long long n_out = 2ll + (long long)N_BAGS * N_BAGS;

    // 1) zero the whole output (A_raw floor: 256MB NT stores)
    k_fill_zero<<<8192, 256, 0, stream>>>(out, n_out);
    // 2) precision-split inputs
    k_split<<<(N_BAGS * L_IN) / 256, 256, 0, stream>>>(x, xhi, xlo, N_BAGS * L_IN);
    k_tsplit<<<(L_IN * H_DIM) / 256, 256, 0, stream>>>(W_fc, wfct_hi, wfct_lo, L_IN, H_DIM);
    // 3) h = relu(x @ W_fc + b)
    k_gemm1<<<dim3(N_BAGS / 128, H_DIM / 128), 256, 0, stream>>>(xhi, xlo, wfct_hi, wfct_lo, b_fc, h);
    // 4) split h; transpose/split gate weights
    k_split<<<(N_BAGS * H_DIM) / 256, 256, 0, stream>>>(h, hhi, hlo, N_BAGS * H_DIM);
    k_tsplit<<<(H_DIM * D_DIM) / 256, 256, 0, stream>>>(W_a, wat_hi, wat_lo, H_DIM, D_DIM);
    k_tsplit<<<(H_DIM * D_DIM) / 256, 256, 0, stream>>>(W_b, wbt_hi, wbt_lo, H_DIM, D_DIM);
    // 5) ab = tanh(h@W_a+b_a) * sigmoid(h@W_b+b_b)
    k_gemm2<<<dim3(N_BAGS / 128, D_DIM / 128), 256, 0, stream>>>(hhi, hlo, wat_hi, wat_lo,
                                                                 wbt_hi, wbt_lo, b_a, b_b, abuf);
    // 6) attention scores, softmax stats, pooled feature, logits, diagonal
    k_scores<<<N_BAGS / 8, 256, 0, stream>>>(abuf, W_c, b_c, scores);
    k_reduce<<<1, 1024, 0, stream>>>(scores, red);
    k_wsum<<<H_DIM / 4, 256, 0, stream>>>(h, scores, red, Mv);
    k_logits<<<1, 256, 0, stream>>>(Mv, W_cls, b_cls, out);
    k_diag<<<N_BAGS / 256, 256, 0, stream>>>(scores, out + 2);
}
